// TransformerMemoryUpdater_69621419868694
// MI455X (gfx1250) — matrix operand order repaired and checked by reading, not compiled, not run on hardware
//
#include <hip/hip_runtime.h>
#include <hip/hip_bf16.h>

// ---------------------------------------------------------------------------
// TransformerMemoryUpdater for MI455X (gfx1250, wave32, WMMA f16->f32)
//   N=32768 nodes, M=10 mails, H=8 heads, DIN=256, DOUT=256, DT=128, DH=32
//   Block = 32 nodes (2 row tiles sharing B fragments), 128 threads = 4 waves.
// ---------------------------------------------------------------------------

#define NT      32      // nodes per block
#define MAILS   10
#define HEADS   8
#define DIN_C   256
#define DT_C    128
#define DX_C    384     // DIN + DT
#define DOUT_C  256
#define EPS_C   1e-5f

typedef __attribute__((ext_vector_type(16))) _Float16 v16h;
typedef __attribute__((ext_vector_type(8)))  float    v8f;

struct alignas(16) h8 { _Float16 h[8]; };
struct alignas(8)  h4 { _Float16 h[4]; };

// B fragment (wave32): lane holds column n = tile*16 + (lane&15),
// K values kstep*32 + hi*8 + {0..7} and + 16 + {0..7} (2x b128 loads).
__device__ __forceinline__ v16h load_b_frag(const _Float16* __restrict__ wT,
                                            int kdim, int col, int kstep, int hi) {
    const _Float16* p = wT + (size_t)col * kdim + kstep * 32 + hi * 8;
    h8 a = *(const h8*)(p);
    h8 b = *(const h8*)(p + 16);
    v16h r;
#pragma unroll
    for (int j = 0; j < 8; ++j) { r[j] = a.h[j]; r[8 + j] = b.h[j]; }
    return r;
}

// A fragment (wave32, 16-bit, 16x32): lane holds row m = lane&15,
// K values = kstep*32 + hi*8 + {0..7} and + 16 + {0..7}.
__device__ __forceinline__ v16h load_a_frag_lds(const _Float16* __restrict__ X,
                                                int stride, int row, int kstep, int hi) {
    const _Float16* p = X + row * stride + kstep * 32 + hi * 8;
    h8 a = *(const h8*)(p);
    h8 b = *(const h8*)(p + 16);
    v16h r;
#pragma unroll
    for (int j = 0; j < 8; ++j) { r[j] = a.h[j]; r[8 + j] = b.h[j]; }
    return r;
}

// A fragment gathered from a global f32 row (for Q = mem @ Wq).
__device__ __forceinline__ v16h load_a_frag_gf32(const float* __restrict__ rowp,
                                                 int kstep, int hi) {
    const float* p = rowp + kstep * 32 + hi * 8;
    v16h r;
#pragma unroll
    for (int j = 0; j < 8; ++j) {
        r[j]     = (_Float16)p[j];
        r[8 + j] = (_Float16)p[16 + j];
    }
    return r;
}

__device__ __forceinline__ v8f wmma_f16(v16h a, v16h b, v8f c) {
    return __builtin_amdgcn_wmma_f32_16x16x32_f16(
        /*neg_a=*/false, a, /*neg_b=*/false, b,
        /*c_mod=*/(short)0, c, /*reuse_a=*/false, /*reuse_b=*/false);
}

__device__ __forceinline__ v8f vzero() {
    v8f z = {0.f, 0.f, 0.f, 0.f, 0.f, 0.f, 0.f, 0.f};
    return z;
}

// ---------------------------------------------------------------------------
// K0: transpose fp32 weight [fin, fout] -> f16 [fout, fin] (contiguous K rows)
// ---------------------------------------------------------------------------
__global__ void transpose_to_f16(const float* __restrict__ src,
                                 _Float16* __restrict__ dst,
                                 int fin, int fout) {
    int idx = blockIdx.x * blockDim.x + threadIdx.x;
    if (idx >= fin * fout) return;
    int o = idx / fin;
    int i = idx - o * fin;
    dst[idx] = (_Float16)src[(size_t)i * fout + o];
}

// ---------------------------------------------------------------------------
// Time-disjoint LDS overlap: phase-3 buffers vs phase-4/5 buffers (48 KB).
// ---------------------------------------------------------------------------
union alignas(16) SmemArena {
    struct {
        _Float16 Xs[NT * DX_C];     // 24 KB: mail feature tile (per mail)
        _Float16 Qs[NT * DOUT_C];   // 16 KB: Q in f16
    } p3;
    struct {
        float    rst[NT * DOUT_C];  // 32 KB: attn out + residual (f32)
        _Float16 Rh[NT * DOUT_C];   // 16 KB: post-LayerNorm (f16)
    } p5;
};

// ---------------------------------------------------------------------------
// Main fused kernel.
// ---------------------------------------------------------------------------
__global__ void __launch_bounds__(128)
tmu_main(const float* __restrict__ mem, const float* __restrict__ mem_input,
         const float* __restrict__ ts, const float* __restrict__ mail_ts,
         const int* __restrict__ node_id,
         const _Float16* __restrict__ wqT, const float* __restrict__ wq_b,
         const _Float16* __restrict__ wkT, const float* __restrict__ wk_b,
         const _Float16* __restrict__ wvT, const float* __restrict__ wv_b,
         const float* __restrict__ ln_g, const float* __restrict__ ln_b,
         const _Float16* __restrict__ mlpT, const float* __restrict__ mlp_b,
         const float* __restrict__ time_w, const float* __restrict__ time_b,
         float* __restrict__ out_rst, int* __restrict__ out_id,
         float* __restrict__ out_ts) {

    __shared__ SmemArena arena;
    __shared__ float ts_s[NT];
    __shared__ float mts_s[NT * MAILS];
    __shared__ float score_s[NT * HEADS];    // 1 KB per-mail scratch
    __shared__ float cs[NT * HEADS];         // rescale factor
    __shared__ float ps[NT * HEADS];         // exp weight
    __shared__ float denom[NT * HEADS];
    __shared__ float mu_s[NT], rsg_s[NT];

    const int tid  = threadIdx.x;            // 0..127
    const int lane = tid & 31;
    const int wv   = tid >> 5;               // wave 0..3
    const int hi   = lane >> 4;              // half-wave
    const int ln16 = lane & 15;
    const int gn0  = blockIdx.x * NT;

    // ---- phase 1: stage timestamps, zero score scratch -----------------
    if (tid < NT) ts_s[tid] = ts[gn0 + tid];
    for (int i = tid; i < NT * MAILS; i += 128)
        mts_s[i] = mail_ts[(size_t)(gn0 + i / MAILS) * MAILS + (i % MAILS)];
    for (int i = tid; i < NT * HEADS; i += 128) score_s[i] = 0.f;
    __syncthreads();

    // ---- phase 2: Q = mem @ Wq + bq (dual row tiles share B) -----------
    for (int t = 0; t < 4; ++t) {
        const int col = (wv * 4 + t) * 16 + ln16;
        v8f acc0 = vzero(), acc1 = vzero();
        const float* arow0 = mem + (size_t)(gn0 + ln16) * DOUT_C;
        const float* arow1 = mem + (size_t)(gn0 + 16 + ln16) * DOUT_C;
#pragma unroll
        for (int ks = 0; ks < 8; ++ks) {
            v16h b  = load_b_frag(wqT, DOUT_C, col, ks, hi);
            v16h a0 = load_a_frag_gf32(arow0, ks, hi);
            v16h a1 = load_a_frag_gf32(arow1, ks, hi);
            acc0 = wmma_f16(a0, b, acc0);
            acc1 = wmma_f16(a1, b, acc1);
        }
        const float qb = wq_b[col];
#pragma unroll
        for (int r = 0; r < 8; ++r) {
            const int n = r + hi * 8;
            arena.p3.Qs[n * DOUT_C + col]        = (_Float16)(acc0[r] + qb);
            arena.p3.Qs[(n + 16) * DOUT_C + col] = (_Float16)(acc1[r] + qb);
        }
    }
    __syncthreads();

    // ---- phase 3: mail loop with online softmax ------------------------
    float vacc[4][2][8];
#pragma unroll
    for (int t = 0; t < 4; ++t)
#pragma unroll
        for (int ro = 0; ro < 2; ++ro)
#pragma unroll
            for (int r = 0; r < 8; ++r) vacc[t][ro][r] = 0.f;

    float wkb[4], wvb[4];
    for (int t = 0; t < 4; ++t) {
        const int col = (wv * 4 + t) * 16 + ln16;
        wkb[t] = wk_b[col];
        wvb[t] = wv_b[col];
    }

    // each thread owns two (n,h) online-softmax states: idx = tid + p*128
    float rm[2] = {-1e30f, -1e30f}, rs[2] = {0.f, 0.f};

    for (int m = 0; m < MAILS; ++m) {
        // build X_m = [mail feats | cos time encoding] in f16
        // mail part: float4-coalesced (NT*64 float4s)
        for (int i = tid; i < NT * (DIN_C / 4); i += 128) {
            const int row = i >> 6;
            const int c4  = i & 63;
            const float4 v = *(const float4*)(mem_input +
                (size_t)(gn0 + row) * (MAILS * DIN_C) + m * DIN_C + c4 * 4);
            h4 hh;
            hh.h[0] = (_Float16)v.x; hh.h[1] = (_Float16)v.y;
            hh.h[2] = (_Float16)v.z; hh.h[3] = (_Float16)v.w;
            *(h4*)(arena.p3.Xs + row * DX_C + c4 * 4) = hh;
        }
        // time-encoding part
        for (int i = tid; i < NT * DT_C; i += 128) {
            const int row = i >> 7;
            const int j   = i & 127;
            const float dt = ts_s[row] - mts_s[row * MAILS + m];
            arena.p3.Xs[row * DX_C + DIN_C + j] =
                (_Float16)__cosf(dt * time_w[j] + time_b[j]);
        }
        __syncthreads();

        // K matmul (K=384 -> 12 ksteps), B shared by both row tiles
        for (int t = 0; t < 4; ++t) {
            const int col = (wv * 4 + t) * 16 + ln16;
            const int h   = col >> 5;
            v8f acc0 = vzero(), acc1 = vzero();
#pragma unroll
            for (int ks = 0; ks < 12; ++ks) {
                v16h b  = load_b_frag(wkT, DX_C, col, ks, hi);
                v16h a0 = load_a_frag_lds(arena.p3.Xs, DX_C, ln16,      ks, hi);
                v16h a1 = load_a_frag_lds(arena.p3.Xs, DX_C, 16 + ln16, ks, hi);
                acc0 = wmma_f16(a0, b, acc0);
                acc1 = wmma_f16(a1, b, acc1);
            }
#pragma unroll
            for (int r = 0; r < 8; ++r) {
                const int n0 = r + hi * 8;
                const int n1 = n0 + 16;
                const float q0 = (float)arena.p3.Qs[n0 * DOUT_C + col];
                const float q1 = (float)arena.p3.Qs[n1 * DOUT_C + col];
                atomicAdd(&score_s[n0 * HEADS + h], q0 * (acc0[r] + wkb[t]));
                atomicAdd(&score_s[n1 * HEADS + h], q1 * (acc1[r] + wkb[t]));
            }
        }
        __syncthreads();

        // leaky-relu + online softmax update; publish c, p; reset scores
#pragma unroll
        for (int p = 0; p < 2; ++p) {
            const int idx = tid + p * 128;
            float sc = score_s[idx];
            sc = sc > 0.f ? sc : 0.2f * sc;
            const float nm = fmaxf(rm[p], sc);
            const float c0 = __expf(rm[p] - nm);
            const float p0 = __expf(sc - nm);
            rs[p] = rs[p] * c0 + p0;
            rm[p] = nm;
            cs[idx] = c0;
            ps[idx] = p0;
            score_s[idx] = 0.f;
        }
        __syncthreads();

        // V matmul on the same X_m, fold into register accumulators
        for (int t = 0; t < 4; ++t) {
            const int col = (wv * 4 + t) * 16 + ln16;
            const int h   = col >> 5;
            v8f acc0 = vzero(), acc1 = vzero();
#pragma unroll
            for (int ks = 0; ks < 12; ++ks) {
                v16h b  = load_b_frag(wvT, DX_C, col, ks, hi);
                v16h a0 = load_a_frag_lds(arena.p3.Xs, DX_C, ln16,      ks, hi);
                v16h a1 = load_a_frag_lds(arena.p3.Xs, DX_C, 16 + ln16, ks, hi);
                acc0 = wmma_f16(a0, b, acc0);
                acc1 = wmma_f16(a1, b, acc1);
            }
#pragma unroll
            for (int r = 0; r < 8; ++r) {
                const int n0 = r + hi * 8;
                const int n1 = n0 + 16;
                vacc[t][0][r] = vacc[t][0][r] * cs[n0 * HEADS + h] +
                                ps[n0 * HEADS + h] * (acc0[r] + wvb[t]);
                vacc[t][1][r] = vacc[t][1][r] * cs[n1 * HEADS + h] +
                                ps[n1 * HEADS + h] * (acc1[r] + wvb[t]);
            }
        }
        __syncthreads();                     // X_m rebuild next iteration
    }

#pragma unroll
    for (int p = 0; p < 2; ++p) denom[tid + p * 128] = rs[p];
    __syncthreads();                         // also: last read of Xs/Qs done

    // ---- phase 4: normalize by softmax denom + residual (arena -> p5) --
    for (int t = 0; t < 4; ++t) {
        const int col = (wv * 4 + t) * 16 + ln16;
        const int h   = col >> 5;
#pragma unroll
        for (int ro = 0; ro < 2; ++ro)
#pragma unroll
            for (int r = 0; r < 8; ++r) {
                const int n = ro * 16 + r + hi * 8;
                const float v = vacc[t][ro][r] / denom[n * HEADS + h] +
                                mem[(size_t)(gn0 + n) * DOUT_C + col];
                arena.p5.rst[n * DOUT_C + col] = v;
            }
    }
    __syncthreads();

    // ---- phase 5: LayerNorm --------------------------------------------
    if (tid < NT) {
        float s = 0.f, s2 = 0.f;
        for (int c = 0; c < DOUT_C; ++c) {
            const float x = arena.p5.rst[tid * DOUT_C + c];
            s += x; s2 += x * x;
        }
        const float mu  = s * (1.f / DOUT_C);
        const float var = s2 * (1.f / DOUT_C) - mu * mu;
        mu_s[tid]  = mu;
        rsg_s[tid] = rsqrtf(var + EPS_C);
    }
    __syncthreads();
    for (int e = tid; e < NT * DOUT_C; e += 128) {
        const int n = e >> 8, c = e & 255;
        const float x = (arena.p5.rst[e] - mu_s[n]) * rsg_s[n] * ln_g[c] + ln_b[c];
        arena.p5.Rh[e] = (_Float16)x;
    }
    __syncthreads();

    // ---- phase 6: MLP + ReLU + store (dual row tiles share B) ----------
    for (int t = 0; t < 4; ++t) {
        const int col = (wv * 4 + t) * 16 + ln16;
        v8f acc0 = vzero(), acc1 = vzero();
#pragma unroll
        for (int ks = 0; ks < 8; ++ks) {
            v16h b  = load_b_frag(mlpT, DOUT_C, col, ks, hi);
            v16h a0 = load_a_frag_lds(arena.p5.Rh, DOUT_C, ln16,      ks, hi);
            v16h a1 = load_a_frag_lds(arena.p5.Rh, DOUT_C, 16 + ln16, ks, hi);
            acc0 = wmma_f16(a0, b, acc0);
            acc1 = wmma_f16(a1, b, acc1);
        }
        const float bb = mlp_b[col];
#pragma unroll
        for (int r = 0; r < 8; ++r) {
            const int n0 = r + hi * 8;
            const int n1 = n0 + 16;
            float v0 = acc0[r] + bb; v0 = v0 > 0.f ? v0 : 0.f;
            float v1 = acc1[r] + bb; v1 = v1 > 0.f ? v1 : 0.f;
            out_rst[(size_t)(gn0 + n0) * DOUT_C + col] = v0;
            out_rst[(size_t)(gn0 + n1) * DOUT_C + col] = v1;
        }
    }

    // ---- tuple tail: node_id (bitwise) and ts --------------------------
    if (tid < NT) {
        out_id[gn0 + tid] = node_id[gn0 + tid];
        out_ts[gn0 + tid] = ts_s[tid];
    }
}

// ---------------------------------------------------------------------------
extern "C" void kernel_launch(void* const* d_in, const int* in_sizes, int n_in,
                              void* d_out, int out_size, void* d_ws, size_t ws_size,
                              hipStream_t stream) {
    const float* mem       = (const float*)d_in[0];
    const float* mem_input = (const float*)d_in[1];
    const float* ts        = (const float*)d_in[2];
    const float* mail_ts   = (const float*)d_in[3];
    const int*   node_id   = (const int*)d_in[4];
    const float* wq_w      = (const float*)d_in[5];
    const float* wq_b      = (const float*)d_in[6];
    const float* wk_w      = (const float*)d_in[7];
    const float* wk_b      = (const float*)d_in[8];
    const float* wv_w      = (const float*)d_in[9];
    const float* wv_b      = (const float*)d_in[10];
    const float* ln_g      = (const float*)d_in[11];
    const float* ln_b      = (const float*)d_in[12];
    const float* mlp_w     = (const float*)d_in[13];
    const float* mlp_b     = (const float*)d_in[14];
    const float* time_w    = (const float*)d_in[15];
    const float* time_b    = (const float*)d_in[16];

    const int Nn = in_sizes[0] / DOUT_C;     // 32768

    // workspace: f16 transposed weights ([out][in], contiguous K)
    _Float16* wqT  = (_Float16*)d_ws;                 // 256*256
    _Float16* wkT  = wqT + 256 * 256;                 // 256*384
    _Float16* wvT  = wkT + 256 * 384;                 // 256*384
    _Float16* mlpT = wvT + 256 * 384;                 // 256*256

    const int T = 256;
    transpose_to_f16<<<(256 * 256 + T - 1) / T, T, 0, stream>>>(wq_w,  wqT,  256, 256);
    transpose_to_f16<<<(384 * 256 + T - 1) / T, T, 0, stream>>>(wk_w,  wkT,  384, 256);
    transpose_to_f16<<<(384 * 256 + T - 1) / T, T, 0, stream>>>(wv_w,  wvT,  384, 256);
    transpose_to_f16<<<(256 * 256 + T - 1) / T, T, 0, stream>>>(mlp_w, mlpT, 256, 256);

    float* out_rst = (float*)d_out;
    int*   out_id  = (int*)(out_rst + (size_t)Nn * DOUT_C);
    float* out_ts  = (float*)(out_id + Nn);

    tmu_main<<<Nn / NT, 128, 0, stream>>>(
        mem, mem_input, ts, mail_ts, node_id,
        wqT, wq_b, wkT, wk_b, wvT, wv_b,
        ln_g, ln_b, mlpT, mlp_b, time_w, time_b,
        out_rst, out_id, out_ts);
}
